// GATLayer_89275190215473
// MI455X (gfx1250) — matrix operand order repaired and checked by reading, not compiled, not run on hardware
//
#include <hip/hip_runtime.h>

#define NODES 20000
#define EDGES 200000
#define INDIM 512
#define LAT 128
#define NH 4
#define OUTC (NH * LAT) /* 512 */

typedef __attribute__((ext_vector_type(16))) __bf16 v16bf;
typedef __attribute__((ext_vector_type(8)))  __bf16 v8bf;
typedef __attribute__((ext_vector_type(8)))  float  v8f;

// Order-preserving float<->uint encoding for atomicMax-based segment max.
__device__ __forceinline__ unsigned enc_f32(float f) {
  unsigned u = __float_as_uint(f);
  return (u & 0x80000000u) ? ~u : (u | 0x80000000u);
}
__device__ __forceinline__ float dec_f32(unsigned u) {
  u = (u & 0x80000000u) ? (u & 0x7FFFFFFFu) : ~u;
  return __uint_as_float(u);
}

// ---------------------------------------------------------------------------
// Kernel 0: repack W (f32 [NH][INDIM][LAT]) into bf16 WMMA B-fragment order.
// Fragment for (tile = hd*8+nt, kstep ks): 32 lanes x 16 bf16, lane-contiguous,
// lane L holds column n = nt*16+(L&15), K = ks*32 + (L>>4)*16 + i  (i=0..15).
// One thread per (tile, ks, lane) slot: 32*16*32 = 16384 threads.
// ---------------------------------------------------------------------------
__global__ __launch_bounds__(256) void pack_w_kernel(const float* __restrict__ W,
                                                     __bf16* __restrict__ Wt) {
  int t    = blockIdx.x * 256 + threadIdx.x;
  int lane = t & 31;
  int ks   = (t >> 5) & 15;
  int tile = t >> 9;                       // 0..31 = hd*8 + nt
  int hd = tile >> 3, nt = tile & 7;
  int n  = nt * 16 + (lane & 15);
  int kb = ks * 32 + (lane >> 4) * 16;
  __bf16* dst = Wt + (size_t)tile * 8192 + ks * 512 + lane * 16;
  const float* src = W + (size_t)hd * INDIM * LAT + n;
#pragma unroll
  for (int i = 0; i < 16; ++i)
    dst[i] = (__bf16)src[(size_t)(kb + i) * LAT];
}

// ---------------------------------------------------------------------------
// Kernel 1: h[n][hd*128+f] = sum_k x[n][k] * W[hd][k][f]  via WMMA bf16.
// One block per 16-row M tile; x tile staged to LDS as bf16 once; 8 waves x
// 4 (head,ntile) output tiles each; 16 WMMA k-steps per tile.
// ---------------------------------------------------------------------------
__global__ __launch_bounds__(256) void gemm_h_kernel(const float* __restrict__ x,
                                                     const __bf16* __restrict__ Wt,
                                                     float* __restrict__ h) {
  __shared__ __bf16 xs[16 * INDIM];        // 16 KB
  const int m0  = blockIdx.x * 16;
  const int tid = threadIdx.x;
  for (int i = tid; i < 16 * INDIM; i += 256)
    xs[i] = (__bf16)x[(size_t)(m0 + (i >> 9)) * INDIM + (i & 511)];
  __syncthreads();

  const int wave = tid >> 5, lane = tid & 31;
  const int lo = lane & 15, hi = lane >> 4;
  // 16-bit A layout: lane L = row M=(L&15); elements 0..15 hold
  // K = (L>>4)*16 + 0..15 (contiguous, matching packed B).
  const __bf16* arow = xs + lo * INDIM;

#pragma unroll
  for (int j = 0; j < 4; ++j) {
    int tile = wave * 4 + j;               // 0..31
    int hd = tile >> 3, nt = tile & 7;
    const __bf16* bt = Wt + (size_t)tile * 8192 + lane * 16;
    v8f c = {};
#pragma unroll
    for (int ks = 0; ks < 16; ++ks) {
      int k0 = ks * 32;
      v8bf a0 = *(const v8bf*)(arow + k0 + hi * 16);       // ds_load_b128
      v8bf a1 = *(const v8bf*)(arow + k0 + 8 + hi * 16);   // ds_load_b128
      v16bf a = __builtin_shufflevector(a0, a1, 0, 1, 2, 3, 4, 5, 6, 7,
                                                8, 9, 10, 11, 12, 13, 14, 15);
      v16bf b = *(const v16bf*)(bt + ks * 512);            // 32B global load
      c = __builtin_amdgcn_wmma_f32_16x16x32_bf16(false, a, false, b,
                                                  (short)0, c, false, false);
    }
    // C/D layout: VGPR r -> M = r + (lane>>4)*8, N = lane&15
    int col = hd * LAT + nt * 16 + lo;
    float* hp = h + (size_t)(m0 + hi * 8) * OUTC + col;
#pragma unroll
    for (int r = 0; r < 8; ++r)
      hp[(size_t)r * OUTC] = c[r];
  }
}

// ---------------------------------------------------------------------------
// Kernel 2: per-node attention scalars alpha_src/alpha_dst (bilinear trick).
// ---------------------------------------------------------------------------
__global__ __launch_bounds__(256) void alpha_kernel(const float* __restrict__ h,
                                                    const float* __restrict__ a,
                                                    float* __restrict__ asrc,
                                                    float* __restrict__ adst) {
  int t = blockIdx.x * 256 + threadIdx.x;
  if (t >= NODES * NH) return;
  int hd = t & 3, n = t >> 2;
  const float* hp = h + (size_t)n * OUTC + hd * LAT;
  const float* ap = a + (size_t)hd * 2 * LAT;
  float s0 = 0.f, s1 = 0.f;
#pragma unroll 8
  for (int f = 0; f < LAT; ++f) {
    float v = hp[f];
    s0 += ap[f] * v;
    s1 += ap[LAT + f] * v;
  }
  asrc[t] = s0;
  adst[t] = s1;
}

// ---------------------------------------------------------------------------
// Kernel 3: init segment max (encoded -inf) and denom.
// ---------------------------------------------------------------------------
__global__ __launch_bounds__(256) void init_seg_kernel(unsigned* __restrict__ segmax,
                                                       float* __restrict__ denom) {
  int t = blockIdx.x * 256 + threadIdx.x;
  if (t < NODES * NH) {
    segmax[t] = 0x007FFFFFu;   // enc_f32(-inf)
    denom[t]  = 0.f;
  }
}

// ---------------------------------------------------------------------------
// Kernel 4: per-edge leaky-relu logits + segment max by src.
// ---------------------------------------------------------------------------
__global__ __launch_bounds__(256) void logits_max_kernel(const int* __restrict__ edges,
                                                         const float* __restrict__ asrc,
                                                         const float* __restrict__ adst,
                                                         float* __restrict__ exbuf,
                                                         unsigned* __restrict__ segmax) {
  int t = blockIdx.x * 256 + threadIdx.x;
  if (t >= EDGES * NH) return;
  int hd = t & 3, e = t >> 2;
  int s = edges[2 * e], d = edges[2 * e + 1];
  float l = asrc[s * NH + hd] + adst[d * NH + hd];
  l = (l > 0.f) ? l : 0.2f * l;            // leaky_relu(0.2)
  exbuf[t] = l;
  atomicMax(&segmax[s * NH + hd], enc_f32(l));
}

// ---------------------------------------------------------------------------
// Kernel 5: ex = exp(logit - segmax[src]); denom[src] += ex.
// ---------------------------------------------------------------------------
__global__ __launch_bounds__(256) void expsum_kernel(const int* __restrict__ edges,
                                                     float* __restrict__ exbuf,
                                                     const unsigned* __restrict__ segmax,
                                                     float* __restrict__ denom) {
  int t = blockIdx.x * 256 + threadIdx.x;
  if (t >= EDGES * NH) return;
  int hd = t & 3, e = t >> 2;
  int s = edges[2 * e];
  float m  = dec_f32(segmax[s * NH + hd]);
  float ex = __expf(exbuf[t] - m);
  exbuf[t] = ex;
  __hip_atomic_fetch_add(&denom[s * NH + hd], ex,
                         __ATOMIC_RELAXED, __HIP_MEMORY_SCOPE_AGENT);
}

// ---------------------------------------------------------------------------
// Kernel 6: out[src] += (ex/denom[src]) * h[dst].  One thread per (edge, col).
// h and out live in L2 (41 MB each << 192 MB), so gathers/atomics stay on-chip.
// ---------------------------------------------------------------------------
__global__ __launch_bounds__(256) void aggregate_kernel(const int* __restrict__ edges,
                                                        const float* __restrict__ exbuf,
                                                        const float* __restrict__ denom,
                                                        const float* __restrict__ h,
                                                        float* __restrict__ out) {
  size_t t = (size_t)blockIdx.x * 256 + threadIdx.x;  // EDGES*512 exact
  int    c = (int)(t & 511);
  size_t e = t >> 9;
  int   hd = c >> 7;
  int s = edges[2 * e], d = edges[2 * e + 1];
  float attn = exbuf[e * NH + hd] / denom[(size_t)s * NH + hd];
  float v = attn * h[(size_t)d * OUTC + c];
  __hip_atomic_fetch_add(&out[(size_t)s * OUTC + c], v,
                         __ATOMIC_RELAXED, __HIP_MEMORY_SCOPE_AGENT);
}

// ---------------------------------------------------------------------------
extern "C" void kernel_launch(void* const* d_in, const int* in_sizes, int n_in,
                              void* d_out, int out_size, void* d_ws, size_t ws_size,
                              hipStream_t stream) {
  const float* x     = (const float*)d_in[0];   // [20000][512]
  const int*   edges = (const int*)d_in[1];     // [200000][2]
  const float* W     = (const float*)d_in[2];   // [4][512][128]
  const float* a     = (const float*)d_in[3];   // [4][256]
  float* out = (float*)d_out;                   // [20000][512]

  // Workspace carve-out (all sizes are multiples of 512 B, so stay aligned).
  char* ws = (char*)d_ws;
  size_t off = 0;
  float*    h     = (float*)(ws + off);   off += (size_t)NODES * OUTC * 4;   // 40.96 MB
  __bf16*   Wt    = (__bf16*)(ws + off);  off += (size_t)32 * 8192 * 2;      // 512 KB
  float*    asrc  = (float*)(ws + off);   off += (size_t)NODES * NH * 4;
  float*    adst  = (float*)(ws + off);   off += (size_t)NODES * NH * 4;
  unsigned* segm  = (unsigned*)(ws + off);off += (size_t)NODES * NH * 4;
  float*    denom = (float*)(ws + off);   off += (size_t)NODES * NH * 4;
  float*    exb   = (float*)(ws + off);   off += (size_t)EDGES * NH * 4;     // 3.2 MB

  // Atomics accumulate into d_out -> must re-zero every call (graph replays).
  hipMemsetAsync(d_out, 0, (size_t)NODES * OUTC * sizeof(float), stream);

  pack_w_kernel<<<16384 / 256, 256, 0, stream>>>(W, Wt);
  gemm_h_kernel<<<NODES / 16, 256, 0, stream>>>(x, Wt, h);
  alpha_kernel<<<(NODES * NH + 255) / 256, 256, 0, stream>>>(h, a, asrc, adst);
  init_seg_kernel<<<(NODES * NH + 255) / 256, 256, 0, stream>>>(segm, denom);
  logits_max_kernel<<<(EDGES * NH + 255) / 256, 256, 0, stream>>>(edges, asrc, adst, exb, segm);
  expsum_kernel<<<(EDGES * NH + 255) / 256, 256, 0, stream>>>(edges, exb, segm, denom);
  aggregate_kernel<<<(unsigned)(((size_t)EDGES * OUTC) / 256), 256, 0, stream>>>(
      edges, exb, denom, h, out);
}
